// MultiHeadSelfAttention_71648644432447
// MI455X (gfx1250) — matrix operand order repaired and checked by reading, not compiled, not run on hardware
//
#include <hip/hip_runtime.h>
#include <hip/hip_bf16.h>

typedef __bf16 bf16_t;
typedef __attribute__((ext_vector_type(16))) __bf16 v16bf;
typedef __attribute__((ext_vector_type(8)))  __bf16 v8bf;
typedef __attribute__((ext_vector_type(8)))  float  v8f;
typedef __attribute__((ext_vector_type(4)))  int    v4i;

#define WMMA_BF16(a, b, c) \
  __builtin_amdgcn_wmma_f32_16x16x32_bf16(false, (a), false, (b), (short)0, (c), false, false)

constexpr int D  = 1024;
constexpr int S  = 4096;
constexpr int BATCH = 2;
constexpr int H  = 16;
constexpr int HD = 64;
constexpr int M_TOTAL = BATCH * S;                 // 8192 rows of (B*S, D)
constexpr size_t MAT_ELEMS = (size_t)M_TOTAL * D;  // 8,388,608

// ---- CDNA5 async global->LDS copy (ASYNCcnt path), feature-gated ----------
#if defined(__has_builtin)
#if __has_builtin(__builtin_amdgcn_global_load_async_to_lds_b128)
#define HAS_ASYNC_LDS 1
#endif
#endif

#ifdef HAS_ASYNC_LDS
typedef __attribute__((address_space(1))) v4i g_v4i;   // clang prints AS1 as __device__
typedef __attribute__((address_space(3))) v4i l_v4i;   // clang prints AS3 as __shared__
__device__ inline void async_copy_b128(const bf16_t* g, bf16_t* l) {
  __builtin_amdgcn_global_load_async_to_lds_b128((g_v4i*)g, (l_v4i*)l, 0, 0);
}
__device__ inline void wait_async0() {
#if __has_builtin(__builtin_amdgcn_s_wait_asynccnt)
  __builtin_amdgcn_s_wait_asynccnt(0);
#else
  asm volatile("s_wait_asynccnt 0x0" ::: "memory");
#endif
}
#endif

// ---- WMMA fragment loaders (contiguous k order, same for A and B, wave32) -------------------
// A (16x32 bf16): lane L -> row L%16; elem e -> K = (L/16)*16 + e
// => two contiguous 8-element (16B) chunks at K offsets half*16 and half*16+8.
__device__ inline v16bf load_a_frag(const bf16_t* row, int half) {
  v8bf lo = *(const v8bf*)(row + half * 16);
  v8bf hi = *(const v8bf*)(row + half * 16 + 8);
  v16bf r;
#pragma unroll
  for (int i = 0; i < 8; i++) { r[i] = lo[i]; r[i + 8] = hi[i]; }
  return r;
}
// B (32x16 bf16): lane L -> col L%16; 16 contiguous K values at K offset
// (L/16)*16  => one contiguous 32B read.
__device__ inline v16bf load_b_frag(const bf16_t* p) {
  return *(const v16bf*)p;
}

// ---- GEMM: out[m][n] = sum_k A[m][k] * W[n][k]   (x @ W^T) ----------------
// MODE 0: f32 A in, bf16 out (V projection)
// MODE 1: f32 A in, bf16 out, RoPE epilogue (Q/K projections)
// MODE 2: bf16 A in, f32 out (final output projection)
// 256 threads (8 waves). Tile 64(M) x 128(N), K-chunk 64; wave: 32x32.
template <int MODE>
__global__ __launch_bounds__(256) void gemm_kernel(const void* __restrict__ Ain,
                                                   const float* __restrict__ W,
                                                   const int* __restrict__ tpos,
                                                   void* __restrict__ Cout) {
  __shared__ __align__(32) bf16_t As[64][64];
  __shared__ __align__(32) bf16_t Bs[128][64];

  const int tid  = threadIdx.x;
  const int lane = tid & 31;
  const int wid  = tid >> 5;
  const int half = lane >> 4;
  const int l16  = lane & 15;
  const int mw   = wid >> 2;                 // 0..1
  const int nw   = wid & 3;                  // 0..3
  const int mBase = blockIdx.x * 64;
  const int nBase = blockIdx.y * 128;

  v8f acc[2][2] = {};

  for (int k0 = 0; k0 < D; k0 += 64) {
    // ---- stage A tile: 64x64, 16 elems/thread ----
    {
      const int idx = tid * 16;
      const int r = idx >> 6, c = idx & 63;
      if (MODE == 2) {
        const bf16_t* A = (const bf16_t*)Ain;
        const bf16_t* p = A + (size_t)(mBase + r) * D + k0 + c;
        *(v16bf*)(&As[r][c]) = *(const v16bf*)p;
        if (k0 + 64 < D) __builtin_prefetch(p + 64, 0, 1);
      } else {
        const float* A = (const float*)Ain;
        const float* p = A + (size_t)(mBase + r) * D + k0 + c;
        v16bf v;
#pragma unroll
        for (int i = 0; i < 16; i++) v[i] = (bf16_t)p[i];
        *(v16bf*)(&As[r][c]) = v;
        if (k0 + 64 < D) __builtin_prefetch(p + 64, 0, 1);
      }
    }
    // ---- stage B tile: 128x64 f32 weights -> bf16, 32 elems/thread ----
    {
      const int idx = tid * 32;
      const int r = idx >> 6, c = idx & 63;        // c in {0, 32}
      const float* p = W + (size_t)(nBase + r) * D + k0 + c;
#pragma unroll
      for (int g = 0; g < 2; g++) {
        v16bf v;
#pragma unroll
        for (int i = 0; i < 16; i++) v[i] = (bf16_t)p[g * 16 + i];
        *(v16bf*)(&Bs[r][c + g * 16]) = v;
      }
      if (k0 + 64 < D) __builtin_prefetch(p + 64, 0, 1);
    }
    __syncthreads();

    v16bf af[2][2], bfr[2][2];
#pragma unroll
    for (int mi = 0; mi < 2; mi++)
#pragma unroll
      for (int ch = 0; ch < 2; ch++)
        af[mi][ch] = load_a_frag(&As[mw * 32 + mi * 16 + l16][ch * 32], half);
#pragma unroll
    for (int ni = 0; ni < 2; ni++)
#pragma unroll
      for (int ch = 0; ch < 2; ch++)
        bfr[ni][ch] = load_b_frag(&Bs[nw * 32 + ni * 16 + l16][ch * 32 + half * 16]);
#pragma unroll
    for (int mi = 0; mi < 2; mi++)
#pragma unroll
      for (int ni = 0; ni < 2; ni++) {
        acc[mi][ni] = WMMA_BF16(af[mi][0], bfr[ni][0], acc[mi][ni]);
        acc[mi][ni] = WMMA_BF16(af[mi][1], bfr[ni][1], acc[mi][ni]);
      }
    __syncthreads();
  }

  // epilogue: (optional RoPE) + store
  // C layout: lane -> col l16; VGPR i -> row half*8 + i
#pragma unroll
  for (int mi = 0; mi < 2; mi++) {
#pragma unroll
    for (int ni = 0; ni < 2; ni++) {
      const int col = nBase + nw * 32 + ni * 16 + l16;
#pragma unroll
      for (int i = 0; i < 8; i++) {
        const int row = mBase + mw * 32 + mi * 16 + half * 8 + i;
        float v = acc[mi][ni][i];
        if (MODE == 1) {
          const int pos = tpos[row & (S - 1)];
          const int d   = col & (HD - 1);
          // inv_freq = theta^{-(d & ~1)/hd};  ln(10000)/64
          const float invf = __expf(-0.14391156831212787f * (float)(d & ~1));
          const float ang  = (float)pos * invf;
          float sn, cs;
          __sincosf(ang, &sn, &cs);
          const float other = __shfl_xor(v, 1, 32);   // paired column
          v = (d & 1) ? (v * cs + other * sn) : (v * cs - other * sn);
        }
        if (MODE == 2) {
          ((float*)Cout)[(size_t)row * D + col] = v;
        } else {
          ((bf16_t*)Cout)[(size_t)row * D + col] = (bf16_t)v;
        }
      }
    }
  }
}

// ---- Flash attention: causal, online softmax ------------------------------
// 128 threads = 4 waves; wave owns a 16-row q tile (workgroup: 64 q rows).
// Key blocks of 64 staged in LDS (K async-copied when available).
__global__ __launch_bounds__(128) void flash_attn_kernel(const bf16_t* __restrict__ Qb,
                                                         const bf16_t* __restrict__ Kb,
                                                         const bf16_t* __restrict__ Vb,
                                                         bf16_t* __restrict__ Ob) {
  __shared__ __align__(32) bf16_t Ks[64][64];      // [key][dim]
  __shared__ __align__(32) bf16_t Vt[64][64];      // [dim][key] (transposed)
  __shared__ __align__(32) bf16_t Ps[4][16][64];   // per-wave P staging

  const int tid  = threadIdx.x;
  const int lane = tid & 31;
  const int wid  = tid >> 5;
  const int half = lane >> 4;
  const int l16  = lane & 15;

  const int bh = blockIdx.x;                 // 0..B*H-1
  const int b  = bh >> 4;
  const int h  = bh & 15;
  const int qBlock = blockIdx.y * 64;
  const int qTile  = qBlock + wid * 16;      // this wave's q rows

  const size_t baseBS = (size_t)b * S;
  const float NEG_INF = -__builtin_inff();

  // Q fragments (A layout), hd=64 => two K=32 chunks, resident whole kernel
  v16bf qf[2];
  {
    const bf16_t* qrow = Qb + (baseBS + qTile + l16) * D + h * HD;
#pragma unroll
    for (int c = 0; c < 2; c++) qf[c] = load_a_frag(qrow + c * 32, half);
  }

  v8f o[4] = {};
  float m_i[8], l_i[8];
#pragma unroll
  for (int i = 0; i < 8; i++) { m_i[i] = NEG_INF; l_i[i] = 0.f; }

  const float scale = 0.125f;                // 1/sqrt(64)
  const int kEnd = qBlock + 64;              // causal bound for workgroup

  for (int kb = 0; kb < kEnd; kb += 64) {
    // ---- stage K block (64x64 bf16): async copy to LDS when available ----
    {
      const int r   = tid >> 1;              // 0..63 (key)
      const int seg = (tid & 1) * 32;        // 32 dims = 64B = 4 x b128
      const bf16_t* kg = Kb + (baseBS + kb + r) * D + h * HD + seg;
      bf16_t* kl = &Ks[r][seg];
#ifdef HAS_ASYNC_LDS
#pragma unroll
      for (int u = 0; u < 4; u++) async_copy_b128(kg + u * 8, kl + u * 8);
#else
#pragma unroll
      for (int u = 0; u < 2; u++)
        *(v16bf*)(kl + u * 16) = *(const v16bf*)(kg + u * 16);
#endif
    }
    // ---- stage V block transposed: packed 2-key b32 stores ----
    {
      const int kp  = tid & 31;              // key pair -> keys {2kp, 2kp+1}
      const int seg = (tid >> 5) * 16;       // 16 dims
      const bf16_t* v0 = Vb + (baseBS + kb + 2 * kp) * D + h * HD + seg;
      v16bf a = *(const v16bf*)v0;
      v16bf bq = *(const v16bf*)(v0 + D);
#pragma unroll
      for (int e = 0; e < 16; e++) {
        unsigned pk = (unsigned)__builtin_bit_cast(unsigned short, a[e]) |
                      ((unsigned)__builtin_bit_cast(unsigned short, bq[e]) << 16);
        *(unsigned*)(&Vt[seg + e][2 * kp]) = pk;
      }
    }
#ifdef HAS_ASYNC_LDS
    wait_async0();
#endif
    __syncthreads();

    if (kb <= qTile + 15) {                  // wave has unmasked work
      // scores: S = Q * K^T  (four 16x16 n-tiles over 64 keys)
      v8f s[4];
#pragma unroll
      for (int j = 0; j < 4; j++) {
        v8f c = {};
#pragma unroll
        for (int ch = 0; ch < 2; ch++) {
          v16bf kfr = load_b_frag(&Ks[j * 16 + l16][ch * 32 + half * 16]);
          c = WMMA_BF16(qf[ch], kfr, c);
        }
        s[j] = c;
      }
      // mask + scale + online softmax (row reductions across 16 lanes/half)
#pragma unroll
      for (int i = 0; i < 8; i++) {
        const int qrow = qTile + half * 8 + i;
#pragma unroll
        for (int j = 0; j < 4; j++) {
          const int key = kb + j * 16 + l16;
          float v = s[j][i] * scale;
          if (key > qrow) v = NEG_INF;
          s[j][i] = v;
        }
        float t = fmaxf(fmaxf(s[0][i], s[1][i]), fmaxf(s[2][i], s[3][i]));
#pragma unroll
        for (int d = 1; d < 16; d <<= 1) t = fmaxf(t, __shfl_xor(t, d, 32));
        const float mn   = fmaxf(m_i[i], t);
        const float corr = __expf(m_i[i] - mn);   // 0 on first block
        m_i[i] = mn;
        float rs = 0.f;
#pragma unroll
        for (int j = 0; j < 4; j++) {
          const float p = __expf(s[j][i] - mn);
          s[j][i] = p;
          rs += p;
        }
#pragma unroll
        for (int d = 1; d < 16; d <<= 1) rs += __shfl_xor(rs, d, 32);
        l_i[i] = l_i[i] * corr + rs;
#pragma unroll
        for (int c = 0; c < 4; c++) o[c][i] *= corr;
      }
      // P: C layout -> A layout via per-wave LDS round trip
#pragma unroll
      for (int i = 0; i < 8; i++) {
        const int r = half * 8 + i;
#pragma unroll
        for (int j = 0; j < 4; j++)
          Ps[wid][r][j * 16 + l16] = (bf16_t)s[j][i];
      }
      v16bf pf[2];
#pragma unroll
      for (int ch = 0; ch < 2; ch++)
        pf[ch] = load_a_frag(&Ps[wid][l16][ch * 32], half);
      // O += P * V  (4 output dim tiles, K=64 in two chunks)
#pragma unroll
      for (int c = 0; c < 4; c++) {
#pragma unroll
        for (int ch = 0; ch < 2; ch++) {
          v16bf vfr = load_b_frag(&Vt[c * 16 + l16][ch * 32 + half * 16]);
          o[c] = WMMA_BF16(pf[ch], vfr, o[c]);
        }
      }
    }
    __syncthreads();
  }

  // normalize and store (B*S, D) with col = h*64+d  == transpose+reshape
#pragma unroll
  for (int i = 0; i < 8; i++) {
    const float inv = 1.0f / l_i[i];
    const int row = qTile + half * 8 + i;
    const size_t base = (baseBS + row) * D + h * HD;
#pragma unroll
    for (int c = 0; c < 4; c++)
      Ob[base + c * 16 + l16] = (bf16_t)(o[c][i] * inv);
  }
}

extern "C" void kernel_launch(void* const* d_in, const int* in_sizes, int n_in,
                              void* d_out, int out_size, void* d_ws, size_t ws_size,
                              hipStream_t stream) {
  const float* x  = (const float*)d_in[0];
  const int* tpos = (const int*)d_in[1];
  const float* Wq = (const float*)d_in[2];
  const float* Wk = (const float*)d_in[3];
  const float* Wv = (const float*)d_in[4];
  const float* Wo = (const float*)d_in[5];
  float* out = (float*)d_out;

  bf16_t* Qb = (bf16_t*)d_ws;
  bf16_t* Kb = Qb + MAT_ELEMS;
  bf16_t* Vb = Kb + MAT_ELEMS;
  bf16_t* Ab = Vb + MAT_ELEMS;   // attention output (bf16), total ws = 64 MB

  dim3 gGrid(M_TOTAL / 64, D / 128), gBlk(256);
  gemm_kernel<1><<<gGrid, gBlk, 0, stream>>>(x, Wq, tpos, Qb);
  gemm_kernel<1><<<gGrid, gBlk, 0, stream>>>(x, Wk, tpos, Kb);
  gemm_kernel<0><<<gGrid, gBlk, 0, stream>>>(x, Wv, tpos, Vb);

  dim3 aGrid(BATCH * H, S / 64), aBlk(128);
  flash_attn_kernel<<<aGrid, aBlk, 0, stream>>>(Qb, Kb, Vb, Ab);

  gemm_kernel<2><<<gGrid, gBlk, 0, stream>>>(Ab, Wo, tpos, out);
}
